// DeformConv2D_77790447665637
// MI455X (gfx1250) — matrix operand order repaired and checked by reading, not compiled, not run on hardware
//
#include <hip/hip_runtime.h>
#include <hip/hip_bf16.h>

// ---------------------------------------------------------------------------
// Deformable Conv2D on MI455X (gfx1250, wave32, WMMA)
//   conv1: x(16,64,128,128) * w_off(128,64,3,3) -> off (fp32 NCHW)
//   deform: bilinear sample of x with off pairs  -> xd (bf16 NHWC padded)
//   conv2: xd * w_conv(64,64,3,3) + b            -> out (fp32 NCHW)
// Both convs run as implicit GEMM on v_wmma_f32_16x16x32_bf16, with packed
// weight B-fragments staged once per workgroup into LDS (ds_load_b128 feed).
// ---------------------------------------------------------------------------

typedef __attribute__((ext_vector_type(16))) __bf16 v16bf;
typedef __attribute__((ext_vector_type(8)))  float  v8f;

#define Bn 16
#define Cc 64
#define Hh 128
#define Ww 128
#define Hp 130
#define Wp 130

__device__ __forceinline__ unsigned short f2bf(float f) {
  // round-to-nearest-even fp32 -> bf16 (bit pattern)
  unsigned int u = __float_as_uint(f);
  return (unsigned short)((u + 0x7FFFu + ((u >> 16) & 1u)) >> 16);
}

// ---------------------------------------------------------------------------
// Pack OIHW fp32 weights into WMMA B-fragments (32x16 bf16 each).
// Fragment index = nt*18 + tap*2 + kc ; within a fragment, element (L,e):
//   column n = L&15, k = (L>>4)*16 + e   (lanes 0-15: K 0-15, lanes 16-31: K 16-31)
// ---------------------------------------------------------------------------
__global__ void repack_w(const float* __restrict__ w, unsigned short* __restrict__ wB,
                         int ntiles) {
  int t = blockIdx.x * blockDim.x + threadIdx.x;
  if (t >= ntiles * 18 * 512) return;
  int e    = t & 15;
  int L    = (t >> 4) & 31;
  int frag = t >> 9;
  int nt   = frag / 18;
  int rem  = frag % 18;
  int tap  = rem >> 1;
  int kc   = rem & 1;
  int kh = tap / 3, kw = tap % 3;
  int n  = L & 15;
  int k  = ((L >> 4) << 4) + e;          // 0..31
  int ic = kc * 32 + k;
  int oc = nt * 16 + n;
  wB[frag * 512 + L * 16 + e] = f2bf(w[((oc * Cc + ic) * 3 + kh) * 3 + kw]);
}

// ---------------------------------------------------------------------------
// NCHW fp32 -> zero-padded NHWC bf16 (B,130,130,64). Channel-contiguous layout
// makes the WMMA A-fragment two b128 loads per lane; padding removes branches.
// ---------------------------------------------------------------------------
__global__ void x_to_nhwc_pad(const float* __restrict__ x, unsigned short* __restrict__ xp) {
  int t = blockIdx.x * blockDim.x + threadIdx.x;
  if (t >= Bn * Cc * Hp * Wp) return;
  int wp = t % Wp;
  int t2 = t / Wp;
  int hp = t2 % Hp;
  int t3 = t2 / Hp;
  int c  = t3 % Cc;
  int b  = t3 / Cc;
  float v = 0.0f;
  if (hp >= 1 && hp <= Hh && wp >= 1 && wp <= Ww)
    v = x[((b * Cc + c) * Hh + (hp - 1)) * Ww + (wp - 1)];
  xp[((b * Hp + hp) * Wp + wp) * Cc + c] = f2bf(v);
}

// ---------------------------------------------------------------------------
// Implicit-GEMM conv via WMMA. One wave owns a 16-pixel row segment and all
// NT*16 output channels: 9 taps x 2 K-chunks x NT  v_wmma_f32_16x16x32_bf16.
// All B-fragments (NT*18 KB) staged into LDS once per workgroup.
// ---------------------------------------------------------------------------
template <int NT, bool HAS_BIAS>
__global__ __launch_bounds__(256)
void conv_wmma(const unsigned short* __restrict__ xin,   // padded NHWC bf16 bits
               const unsigned short* __restrict__ wB,    // packed B fragments (global)
               const float* __restrict__ bias,
               float* __restrict__ out) {                // NCHW fp32
  extern __shared__ char smem_raw[];
  unsigned short* sB = (unsigned short*)smem_raw;

  // Cooperative stage of all NT*18 B-fragments (1 KB each) into LDS.
  {
    const uint4* gsrc = (const uint4*)wB;
    uint4*       sdst = (uint4*)smem_raw;
    const int nvec = NT * 18 * 1024 / 16;
#pragma unroll 4
    for (int i = threadIdx.x; i < nvec; i += 256) sdst[i] = gsrc[i];
  }
  __syncthreads();

  int tid  = blockIdx.x * blockDim.x + threadIdx.x;
  int wave = tid >> 5;
  int lane = tid & 31;
  if (wave >= Bn * Hh * (Ww / 16)) return;   // grid is exact; formality
  int wt = wave & 7;            // 16-pixel tile along W
  int h  = (wave >> 3) & 127;
  int b  = wave >> 10;
  int m  = lane & 15;           // pixel within tile (A row) / output channel (D col)
  int hi = lane >> 4;

  v8f acc[NT];
#pragma unroll
  for (int i = 0; i < NT; ++i) acc[i] = (v8f)(0.0f);

  union AB { uint4 q[2]; v16bf v; };

#pragma unroll
  for (int tap = 0; tap < 9; ++tap) {
    const int kh = tap / 3, kw = tap % 3;
    const int row = h + kh;                       // padded row
    const int col = wt * 16 + kw + m;             // padded col for this lane
    const unsigned short* abase = xin + ((size_t)(b * Hp + row) * Wp + col) * Cc;
#pragma unroll
    for (int kc = 0; kc < 2; ++kc) {
      // A 16x32 bf16: lane(hi) elems 0-15 -> K = hi*16+e
      AB a;
      a.q[0] = *(const uint4*)(abase + kc * 32 + hi * 16);
      a.q[1] = *(const uint4*)(abase + kc * 32 + hi * 16 + 8);
#pragma unroll
      for (int nt = 0; nt < NT; ++nt) {
        AB bf;
        const uint4* pb = (const uint4*)(sB + ((nt * 9 + tap) * 2 + kc) * 512 + lane * 16);
        bf.q[0] = pb[0];                          // ds_load_b128
        bf.q[1] = pb[1];                          // ds_load_b128
        acc[nt] = __builtin_amdgcn_wmma_f32_16x16x32_bf16(
            false, a.v, false, bf.v, (short)0, acc[nt], false, false);
      }
    }
  }

  // D layout: VGPR r -> (M = hi*8 + r, N = lane&15): 8 floats contiguous in W.
  const int OC = NT * 16;
#pragma unroll
  for (int nt = 0; nt < NT; ++nt) {
    int n = nt * 16 + m;
    float bv = HAS_BIAS ? bias[n] : 0.0f;
    float* po = out + ((size_t)(b * OC + n) * Hh + h) * Ww + wt * 16 + hi * 8;
    float4 s0 = make_float4(acc[nt][0] + bv, acc[nt][1] + bv, acc[nt][2] + bv, acc[nt][3] + bv);
    float4 s1 = make_float4(acc[nt][4] + bv, acc[nt][5] + bv, acc[nt][6] + bv, acc[nt][7] + bv);
    *(float4*)(po)     = s0;
    *(float4*)(po + 4) = s1;
  }
}

// ---------------------------------------------------------------------------
// Bilinear deform sampling. The torch-style .view(-1,H,W,2) reinterpretation
// means (row_off, col_off) for (b,c,hw) are 2 consecutive floats at flat
// offset 2*hw inside the 2-channel slab starting at channel 2c of batch b.
// Writes bf16 into padded NHWC staging for conv2 (borders -> 0).
// ---------------------------------------------------------------------------
__global__ void deform(const float* __restrict__ x, const float* __restrict__ off,
                       unsigned short* __restrict__ xdp) {
  int t = blockIdx.x * blockDim.x + threadIdx.x;
  if (t >= Bn * Hp * Wp * Cc) return;
  int c  = t & 63;
  int t2 = t >> 6;
  int wp = t2 % Wp;
  int t3 = t2 / Wp;
  int hp = t3 % Hp;
  int b  = t3 / Hp;
  unsigned short res = 0;
  if (hp >= 1 && hp <= Hh && wp >= 1 && wp <= Ww) {
    int h = hp - 1, w = wp - 1;
    int hw = h * Ww + w;
    const float2 o2 = *(const float2*)(off + (size_t)(b * 128 + 2 * c) * (Hh * Ww) + 2 * hw);
    float rows = fminf(fmaxf(o2.x + (float)h, 0.0f), 127.0f);
    float cols = fminf(fmaxf(o2.y + (float)w, 0.0f), 127.0f);
    float r0 = floorf(rows), r1 = ceilf(rows);
    float c0 = floorf(cols), c1 = ceilf(cols);
    const float* xp = x + (size_t)(b * Cc + c) * (Hh * Ww);
    int ir0 = (int)r0, ir1 = (int)r1, ic0 = (int)c0, ic1 = (int)c1;
    float v_lt = xp[ir0 * Ww + ic0];
    float v_rb = xp[ir1 * Ww + ic1];
    float v_lb = xp[ir0 * Ww + ic1];
    float v_rt = xp[ir1 * Ww + ic0];
    float fr = rows - r0, fc = cols - c0;
    float vt = fr * (v_rt - v_lt) + v_lt;
    float vb = fr * (v_rb - v_lb) + v_lb;
    res = f2bf(fc * (vb - vt) + vt);
  }
  xdp[((size_t)(b * Hp + hp) * Wp + wp) * Cc + c] = res;
}

// ---------------------------------------------------------------------------
extern "C" void kernel_launch(void* const* d_in, const int* in_sizes, int n_in,
                              void* d_out, int out_size, void* d_ws, size_t ws_size,
                              hipStream_t stream) {
  const float* x      = (const float*)d_in[0];
  const float* w_off  = (const float*)d_in[1];
  const float* w_conv = (const float*)d_in[2];
  const float* b_conv = (const float*)d_in[3];
  float* out = (float*)d_out;

  // Workspace carve-up (~204 MB total)
  char* ws = (char*)d_ws;
  const size_t padElems = (size_t)Bn * Hp * Wp * Cc;          // 17,305,600
  const size_t offElems = (size_t)Bn * 128 * Hh * Ww;         // 33,554,432
  unsigned short* x_pad  = (unsigned short*)ws;
  unsigned short* xd_pad = x_pad + padElems;
  float*          off    = (float*)(ws + 2 * padElems * sizeof(unsigned short));
  unsigned short* wB1    = (unsigned short*)((char*)off + offElems * sizeof(float));
  unsigned short* wB2    = wB1 + 8 * 18 * 512;

  // 1) weight packing (tiny)
  repack_w<<<(8 * 18 * 512 + 255) / 256, 256, 0, stream>>>(w_off, wB1, 8);
  repack_w<<<(4 * 18 * 512 + 255) / 256, 256, 0, stream>>>(w_conv, wB2, 4);

  // 2) stage x -> padded NHWC bf16
  x_to_nhwc_pad<<<(int)((padElems + 255) / 256), 256, 0, stream>>>(x, x_pad);

  // 3) offset conv (64 -> 128 ch), WMMA, B-fragments in LDS (144 KB)
  const int convBlocks = (Bn * Hh * (Ww / 16) * 32) / 256;    // 2048
  conv_wmma<8, false><<<convBlocks, 256, 8 * 18 * 1024, stream>>>(x_pad, wB1, nullptr, off);

  // 4) deformable bilinear sampling -> padded NHWC bf16
  deform<<<(int)((padElems + 255) / 256), 256, 0, stream>>>(x, off, xd_pad);

  // 5) main conv (64 -> 64 ch) + bias, WMMA, B-fragments in LDS (72 KB)
  conv_wmma<4, true><<<convBlocks, 256, 4 * 18 * 1024, stream>>>(xd_pad, wB2, b_conv, out);
}
